// VideoEncDec_90529320665985
// MI455X (gfx1250) — matrix operand order repaired and checked by reading, not compiled, not run on hardware
//
#include <hip/hip_runtime.h>
#include <math.h>

// ---------------------------------------------------------------- constants
constexpr int BS   = 64;
constexpr int SEQ  = 300;
constexpr int D    = 768;
constexpr int D3   = 3 * D;     // 2304
constexpr int TMAX = 100;

// ---------------------------------------------------------------- WMMA types
typedef __attribute__((ext_vector_type(16))) __bf16 v16bf;
typedef __attribute__((ext_vector_type(8)))  __bf16 v8bf;
typedef __attribute__((ext_vector_type(8)))  float  v8f;

// A-fragment (16-bit A 16x32): lane holds row (lane&15),
// K contiguous [o,o+16), o = (lane>=16)?16:0.
__device__ __forceinline__ v16bf load_a_frag(const __bf16* base) {
  v8bf lo = *(const v8bf*)(base);
  v8bf hi = *(const v8bf*)(base + 8);
  return __builtin_shufflevector(lo, hi, 0,1,2,3,4,5,6,7,8,9,10,11,12,13,14,15);
}

__device__ __forceinline__ v8f wmma_bf16(v16bf a, v16bf b, v8f c) {
  return __builtin_amdgcn_wmma_f32_16x16x32_bf16(false, a, false, b, (short)0, c, false, false);
}

__device__ __forceinline__ float sigmoidf_(float x) { return 1.0f / (1.0f + expf(-x)); }

// ---------------------------------------------------------------- big GEMM
// C[M,N] = A[M,K] @ B[N,K]^T + bias[N]
// A, B bf16 row-major; C fp32. M%16==0, N%64==0, K%32==0.
// One wave computes a 16x64 C tile (A fragment reused across 4 B tiles).
__global__ void __launch_bounds__(256)
wmma_gemm_bf16(const __bf16* __restrict__ A, const __bf16* __restrict__ B,
               const float* __restrict__ bias, float* __restrict__ C,
               int M, int N, int K) {
  const int gwave  = (int)((blockIdx.x * blockDim.x + threadIdx.x) >> 5);
  const int lane   = (int)(threadIdx.x & 31);
  const int tilesN = N >> 6;
  const int nwaves = (M >> 4) * tilesN;
  if (gwave >= nwaves) return;               // whole-wave uniform exit
  const int tm  = gwave / tilesN;
  const int tn  = gwave - tm * tilesN;
  const int l15 = lane & 15;
  const int hiL = lane >> 4;
  const int row = (tm << 4) + l15;
  const int col = (tn << 6) + l15;

  const __bf16* Ap = A + (size_t)row * K + hiL * 16;
  const __bf16* B0 = B + (size_t)(col +  0) * K + hiL * 16;
  const __bf16* B1 = B + (size_t)(col + 16) * K + hiL * 16;
  const __bf16* B2 = B + (size_t)(col + 32) * K + hiL * 16;
  const __bf16* B3 = B + (size_t)(col + 48) * K + hiL * 16;

  v8f acc0 = {}, acc1 = {}, acc2 = {}, acc3 = {};
  for (int k = 0; k < K; k += 32) {
    v16bf a  = load_a_frag(Ap + k);
    v16bf b0 = *(const v16bf*)(B0 + k);
    v16bf b1 = *(const v16bf*)(B1 + k);
    v16bf b2 = *(const v16bf*)(B2 + k);
    v16bf b3 = *(const v16bf*)(B3 + k);
    acc0 = wmma_bf16(a, b0, acc0);
    acc1 = wmma_bf16(a, b1, acc1);
    acc2 = wmma_bf16(a, b2, acc2);
    acc3 = wmma_bf16(a, b3, acc3);
  }

  const int rbase = (tm << 4) + hiL * 8;     // C/D: VGPR i -> row i (+8 hi half)
  const int c0    = (tn << 6) + l15;
  const float bb0 = bias[c0], bb1 = bias[c0 + 16], bb2 = bias[c0 + 32], bb3 = bias[c0 + 48];
#pragma unroll
  for (int i = 0; i < 8; ++i) {
    size_t r = (size_t)(rbase + i) * N;
    C[r + c0 +  0] = acc0[i] + bb0;
    C[r + c0 + 16] = acc1[i] + bb1;
    C[r + c0 + 32] = acc2[i] + bb2;
    C[r + c0 + 48] = acc3[i] + bb3;
  }
}

// ---------------------------------------------------------------- fused GRU steps
// Encoder step: gh = h @ Whh^T (WMMA, 3 gate accumulators per 16x16 j-tile),
// then gate combine fused in epilogue. Double-buffered h to avoid RAW races.
// Grid: exactly (BS/16)*(D/16) = 192 waves = 24 blocks x 256.
__global__ void __launch_bounds__(256)
gru_step_enc(const __bf16* __restrict__ hbf_in, const __bf16* __restrict__ Whhb,
             const float* __restrict__ gi_base,   // (BS, 3D) fp32, bih included
             const float* __restrict__ bhh,
             const float* __restrict__ h_in, float* __restrict__ h_out,
             __bf16* __restrict__ hbf_out,
             float* __restrict__ enc_out, __bf16* __restrict__ enc_bf, int step) {
  const int gwave = (int)((blockIdx.x * blockDim.x + threadIdx.x) >> 5);  // 0..191
  const int lane  = (int)(threadIdx.x & 31);
  const int tm  = gwave / (D / 16);        // 0..3  (batch tile)
  const int tj  = gwave - tm * (D / 16);   // 0..47 (hidden tile)
  const int l15 = lane & 15;
  const int hiL = lane >> 4;
  const int row  = tm * 16 + l15;          // batch index for A
  const int colj = tj * 16 + l15;          // hidden index for B

  const __bf16* Ap = hbf_in + (size_t)row * D + hiL * 16;
  const __bf16* Br = Whhb + (size_t)(colj)         * D + hiL * 16;
  const __bf16* Bz = Whhb + (size_t)(D + colj)     * D + hiL * 16;
  const __bf16* Bn = Whhb + (size_t)(2 * D + colj) * D + hiL * 16;

  v8f aR = {}, aZ = {}, aN = {};
  for (int k = 0; k < D; k += 32) {
    v16bf a  = load_a_frag(Ap + k);
    v16bf br = *(const v16bf*)(Br + k);
    v16bf bz = *(const v16bf*)(Bz + k);
    v16bf bn = *(const v16bf*)(Bn + k);
    aR = wmma_bf16(a, br, aR);
    aZ = wmma_bf16(a, bz, aZ);
    aN = wmma_bf16(a, bn, aN);
  }

  const int b0 = tm * 16 + hiL * 8;
  const int j  = tj * 16 + l15;
  const float br_ = bhh[j], bz_ = bhh[D + j], bn_ = bhh[2 * D + j];
#pragma unroll
  for (int i = 0; i < 8; ++i) {
    int b = b0 + i;
    size_t gbase = (size_t)b * D3 + j;
    float r = sigmoidf_(gi_base[gbase]         + aR[i] + br_);
    float z = sigmoidf_(gi_base[gbase + D]     + aZ[i] + bz_);
    float n = tanhf   (gi_base[gbase + 2 * D]  + r * (aN[i] + bn_));
    float hn = (1.0f - z) * n + z * h_in[(size_t)b * D + j];
    __bf16 hb = (__bf16)hn;
    h_out[(size_t)b * D + j]   = hn;
    hbf_out[(size_t)b * D + j] = hb;
    size_t eo = ((size_t)b * SEQ + step) * D + j;
    enc_out[eo] = hn;
    enc_bf[eo]  = hb;
  }
}

// Decoder step: gi = ctx @ W1^T, gh = h @ Whh^T (6 accumulators), plus the
// rank-1 last*dec_Wih[:,0] term and both biases, gate combine in epilogue.
__global__ void __launch_bounds__(256)
gru_step_dec(const __bf16* __restrict__ hbf_in, const __bf16* __restrict__ ctxbf,
             const __bf16* __restrict__ Whhb, const __bf16* __restrict__ W1b,
             const float* __restrict__ Wih,   // (3D, 1+D) fp32 (for column 0)
             const float* __restrict__ bih, const float* __restrict__ bhh,
             const float* __restrict__ last,
             const float* __restrict__ h_in, float* __restrict__ h_out,
             __bf16* __restrict__ hbf_out) {
  const int gwave = (int)((blockIdx.x * blockDim.x + threadIdx.x) >> 5);  // 0..191
  const int lane  = (int)(threadIdx.x & 31);
  const int tm  = gwave / (D / 16);
  const int tj  = gwave - tm * (D / 16);
  const int l15 = lane & 15;
  const int hiL = lane >> 4;
  const int row  = tm * 16 + l15;
  const int colj = tj * 16 + l15;

  const __bf16* Ah = hbf_in + (size_t)row * D + hiL * 16;
  const __bf16* Ac = ctxbf  + (size_t)row * D + hiL * 16;
  const __bf16* Hr = Whhb + (size_t)(colj)         * D + hiL * 16;
  const __bf16* Hz = Whhb + (size_t)(D + colj)     * D + hiL * 16;
  const __bf16* Hn = Whhb + (size_t)(2 * D + colj) * D + hiL * 16;
  const __bf16* Ir = W1b  + (size_t)(colj)         * D + hiL * 16;
  const __bf16* Iz = W1b  + (size_t)(D + colj)     * D + hiL * 16;
  const __bf16* In = W1b  + (size_t)(2 * D + colj) * D + hiL * 16;

  v8f ghR = {}, ghZ = {}, ghN = {}, giR = {}, giZ = {}, giN = {};
  for (int k = 0; k < D; k += 32) {
    v16bf ah = load_a_frag(Ah + k);
    v16bf ac = load_a_frag(Ac + k);
    v16bf hr = *(const v16bf*)(Hr + k);
    v16bf hz = *(const v16bf*)(Hz + k);
    v16bf hn = *(const v16bf*)(Hn + k);
    v16bf ir = *(const v16bf*)(Ir + k);
    v16bf iz = *(const v16bf*)(Iz + k);
    v16bf in = *(const v16bf*)(In + k);
    ghR = wmma_bf16(ah, hr, ghR);
    ghZ = wmma_bf16(ah, hz, ghZ);
    ghN = wmma_bf16(ah, hn, ghN);
    giR = wmma_bf16(ac, ir, giR);
    giZ = wmma_bf16(ac, iz, giZ);
    giN = wmma_bf16(ac, in, giN);
  }

  const int b0 = tm * 16 + hiL * 8;
  const int j  = tj * 16 + l15;
  const float bihr = bih[j], bihz = bih[D + j], bihn = bih[2 * D + j];
  const float bhhr = bhh[j], bhhz = bhh[D + j], bhhn = bhh[2 * D + j];
  const float w0r = Wih[(size_t)j * (D + 1)];
  const float w0z = Wih[(size_t)(D + j) * (D + 1)];
  const float w0n = Wih[(size_t)(2 * D + j) * (D + 1)];
#pragma unroll
  for (int i = 0; i < 8; ++i) {
    int b = b0 + i;
    float l = last[b];
    float r = sigmoidf_((giR[i] + bihr + l * w0r) + (ghR[i] + bhhr));
    float z = sigmoidf_((giZ[i] + bihz + l * w0z) + (ghZ[i] + bhhz));
    float n = tanhf    ((giN[i] + bihn + l * w0n) + r * (ghN[i] + bhhn));
    float hn = (1.0f - z) * n + z * h_in[(size_t)b * D + j];
    h_out[(size_t)b * D + j]   = hn;
    hbf_out[(size_t)b * D + j] = (__bf16)hn;
  }
}

// ---------------------------------------------------------------- helpers
__global__ void f32_to_bf16(const float* __restrict__ in, __bf16* __restrict__ out, int n) {
  int i = blockIdx.x * blockDim.x + threadIdx.x;
  if (i < n) out[i] = (__bf16)in[i];
}

// dec_Wih is (3d, 1+d); strip column 0 -> bf16 (3d, d)
__global__ void slice_dec_wih(const float* __restrict__ in, __bf16* __restrict__ out) {
  int i = blockIdx.x * blockDim.x + threadIdx.x;
  if (i >= D3 * D) return;
  int g = i / D, k = i - g * D;
  out[i] = (__bf16)in[(size_t)g * (D + 1) + 1 + k];
}

// Build A for gi_all GEMM: rows ordered (s*BS + b); value = x[b,s,:] + PE[s,:]
__global__ void build_xpe(const float* __restrict__ x, __bf16* __restrict__ xpe) {
  size_t idx = (size_t)blockIdx.x * blockDim.x + threadIdx.x;
  if (idx >= (size_t)SEQ * BS * D) return;
  int j = (int)(idx % D);
  size_t t = idx / D;
  int b = (int)(t % BS);
  int s = (int)(t / BS);
  float div = expf(-(float)(j & ~1) * (logf(10000.0f) / (float)D));
  float ang = (float)s * div;
  float pe  = (j & 1) ? cosf(ang) : sinf(ang);
  xpe[idx] = (__bf16)(x[((size_t)b * SEQ + s) * D + j] + pe);
}

__global__ void init_state(float* h, __bf16* hbf, float* last) {
  int i = blockIdx.x * blockDim.x + threadIdx.x;
  if (i < BS * D) { h[i] = 0.0f; hbf[i] = (__bf16)0.0f; }
  if (i < BS)     last[i] = 0.0f;
}

// logits[b,s] = tanh(mem[b,s,:] + q[b,:]) . pW + pb   (one wave per logit)
__global__ void logits_kernel(const float* __restrict__ mem, const float* __restrict__ q,
                              const float* __restrict__ pW, const float* __restrict__ pb,
                              float* __restrict__ logits) {
  int b = blockIdx.x / (SEQ / 4);
  int s = (blockIdx.x % (SEQ / 4)) * 4 + (threadIdx.x >> 5);
  int lane = threadIdx.x & 31;
  const float* mp = mem + ((size_t)b * SEQ + s) * D;
  const float* qp = q + (size_t)b * D;
  float acc = 0.0f;
  for (int j = lane; j < D; j += 32) acc += tanhf(mp[j] + qp[j]) * pW[j];
  for (int off = 16; off; off >>= 1) acc += __shfl_down(acc, off, 32);
  if (lane == 0) logits[b * SEQ + s] = acc + pb[0];
}

__global__ void softmax_kernel(const float* __restrict__ logits, float* __restrict__ w) {
  __shared__ float red[256];
  int b = blockIdx.x, tid = threadIdx.x;
  float m = -1e30f;
  for (int s = tid; s < SEQ; s += 256) m = fmaxf(m, logits[b * SEQ + s]);
  red[tid] = m; __syncthreads();
  for (int off = 128; off; off >>= 1) { if (tid < off) red[tid] = fmaxf(red[tid], red[tid + off]); __syncthreads(); }
  m = red[0]; __syncthreads();
  float sum = 0.0f;
  for (int s = tid; s < SEQ; s += 256) { float e = expf(logits[b * SEQ + s] - m); w[b * SEQ + s] = e; sum += e; }
  red[tid] = sum; __syncthreads();
  for (int off = 128; off; off >>= 1) { if (tid < off) red[tid] += red[tid + off]; __syncthreads(); }
  float inv = 1.0f / red[0];
  for (int s = tid; s < SEQ; s += 256) w[b * SEQ + s] *= inv;
}

// ctx[b,j] = sum_s w[b,s] * enc[b,s,j]  -> bf16 (feeds WMMA A of gi GEMM)
__global__ void ctx_kernel(const float* __restrict__ w, const float* __restrict__ enc,
                           __bf16* __restrict__ ctxbf) {
  int idx = blockIdx.x * blockDim.x + threadIdx.x;
  if (idx >= BS * D) return;
  int b = idx / D, j = idx - b * D;
  const float* wp = w + (size_t)b * SEQ;
  const float* ep = enc + (size_t)b * SEQ * D + j;
  float acc = 0.0f;
  for (int s = 0; s < SEQ; ++s) acc += wp[s] * ep[(size_t)s * D];
  ctxbf[idx] = (__bf16)acc;
}

// pred[b] = h[b,:] . tW + tb ; writes output[b, t] and last[b]
__global__ void pred_kernel(const float* __restrict__ h, const float* __restrict__ tW,
                            const float* __restrict__ tb, float* __restrict__ out,
                            float* __restrict__ last, int t) {
  __shared__ float red[256];
  int b = blockIdx.x, tid = threadIdx.x;
  float acc = 0.0f;
  for (int j = tid; j < D; j += 256) acc += h[(size_t)b * D + j] * tW[j];
  red[tid] = acc; __syncthreads();
  for (int off = 128; off; off >>= 1) { if (tid < off) red[tid] += red[tid + off]; __syncthreads(); }
  if (tid == 0) { float p = red[0] + tb[0]; out[(size_t)b * TMAX + t] = p; last[b] = p; }
}

// ---------------------------------------------------------------- launch
static inline int gemm_blocks(int M, int N) {
  int waves = (M / 16) * (N / 64);
  return (waves * 32 + 255) / 256;
}

extern "C" void kernel_launch(void* const* d_in, const int* in_sizes, int n_in,
                              void* d_out, int out_size, void* d_ws, size_t ws_size,
                              hipStream_t stream) {
  const float* x        = (const float*)d_in[0];
  const float* enc_Wih  = (const float*)d_in[1];
  const float* enc_Whh  = (const float*)d_in[2];
  const float* enc_bih  = (const float*)d_in[3];
  const float* enc_bhh  = (const float*)d_in[4];
  const float* dec_Wih  = (const float*)d_in[5];
  const float* dec_Whh  = (const float*)d_in[6];
  const float* dec_bih  = (const float*)d_in[7];
  const float* dec_bhh  = (const float*)d_in[8];
  const float* qW       = (const float*)d_in[9];
  const float* qb       = (const float*)d_in[10];
  const float* mW       = (const float*)d_in[11];
  const float* mb       = (const float*)d_in[12];
  const float* pW       = (const float*)d_in[13];
  const float* pb       = (const float*)d_in[14];
  const float* tW       = (const float*)d_in[15];
  const float* tb       = (const float*)d_in[16];
  float* out = (float*)d_out;

  // ---- workspace carve (256B aligned)
  char* p = (char*)d_ws;
  auto alloc = [&](size_t bytes) -> void* {
    void* r = (void*)p;
    p += (bytes + 255) & ~(size_t)255;
    return r;
  };
  const size_t TOK = (size_t)SEQ * BS;                 // 19200 rows
  float*  gi_all   = (float*) alloc(TOK * D3 * 4);     // 177 MB
  __bf16* xpe      = (__bf16*)alloc(TOK * D * 2);
  float*  enc_out  = (float*) alloc(TOK * D * 4);
  __bf16* enc_bf   = (__bf16*)alloc(TOK * D * 2);
  float*  mem      = (float*) alloc(TOK * D * 4);
  __bf16* encWihB  = (__bf16*)alloc((size_t)D3 * D * 2);
  __bf16* encWhhB  = (__bf16*)alloc((size_t)D3 * D * 2);
  __bf16* decW1B   = (__bf16*)alloc((size_t)D3 * D * 2);
  __bf16* decWhhB  = (__bf16*)alloc((size_t)D3 * D * 2);
  __bf16* qWB      = (__bf16*)alloc((size_t)D * D * 2);
  __bf16* mWB      = (__bf16*)alloc((size_t)D * D * 2);
  float*  h0       = (float*) alloc((size_t)BS * D * 4);   // double-buffered h
  float*  h1       = (float*) alloc((size_t)BS * D * 4);
  __bf16* hb0      = (__bf16*)alloc((size_t)BS * D * 2);
  __bf16* hb1      = (__bf16*)alloc((size_t)BS * D * 2);
  float*  qbuf     = (float*) alloc((size_t)BS * D * 4);
  float*  logitsb  = (float*) alloc((size_t)BS * SEQ * 4);
  float*  wbuf     = (float*) alloc((size_t)BS * SEQ * 4);
  __bf16* ctxbf    = (__bf16*)alloc((size_t)BS * D * 2);
  float*  last     = (float*) alloc((size_t)BS * 4);

  const int B256 = 256;
  auto gb = [](size_t n) { return (int)((n + 255) / 256); };

  // ---- weight conversions (deterministic each call)
  f32_to_bf16<<<gb((size_t)D3 * D), B256, 0, stream>>>(enc_Wih, encWihB, D3 * D);
  f32_to_bf16<<<gb((size_t)D3 * D), B256, 0, stream>>>(enc_Whh, encWhhB, D3 * D);
  f32_to_bf16<<<gb((size_t)D3 * D), B256, 0, stream>>>(dec_Whh, decWhhB, D3 * D);
  f32_to_bf16<<<gb((size_t)D  * D), B256, 0, stream>>>(qW, qWB, D * D);
  f32_to_bf16<<<gb((size_t)D  * D), B256, 0, stream>>>(mW, mWB, D * D);
  slice_dec_wih<<<gb((size_t)D3 * D), B256, 0, stream>>>(dec_Wih, decW1B);
  build_xpe<<<gb(TOK * D), B256, 0, stream>>>(x, xpe);

  float*  h_cur  = h0;  float*  h_nxt  = h1;
  __bf16* hb_cur = hb0; __bf16* hb_nxt = hb1;
  init_state<<<gb((size_t)BS * D), B256, 0, stream>>>(h_cur, hb_cur, last);

  // ---- gi_all = (x+PE) @ enc_Wih^T + enc_bih   (19200 x 2304 x 768)
  wmma_gemm_bf16<<<gemm_blocks(SEQ * BS, D3), B256, 0, stream>>>(
      xpe, encWihB, enc_bih, gi_all, SEQ * BS, D3, D);

  // ---- encoder scan: 300 sequential fused steps (1 launch/step)
  const int GRU_BLOCKS = (BS / 16) * (D / 16) * 32 / B256;   // 24
  for (int step = 0; step < SEQ; ++step) {
    gru_step_enc<<<GRU_BLOCKS, B256, 0, stream>>>(
        hb_cur, encWhhB, gi_all + (size_t)step * BS * D3, enc_bhh,
        h_cur, h_nxt, hb_nxt, enc_out, enc_bf, step);
    float* tf = h_cur;  h_cur = h_nxt;  h_nxt = tf;
    __bf16* tb16 = hb_cur; hb_cur = hb_nxt; hb_nxt = tb16;
  }

  // ---- mem = enc_outputs @ mW^T + mb   (19200 x 768 x 768)
  wmma_gemm_bf16<<<gemm_blocks(SEQ * BS, D), B256, 0, stream>>>(
      enc_bf, mWB, mb, mem, SEQ * BS, D, D);

  // ---- decoder: 100 sequential steps
  for (int t = 0; t < TMAX; ++t) {
    wmma_gemm_bf16<<<gemm_blocks(BS, D), B256, 0, stream>>>(
        hb_cur, qWB, qb, qbuf, BS, D, D);                    // q = h @ qW^T + qb
    logits_kernel<<<BS * (SEQ / 4), 128, 0, stream>>>(mem, qbuf, pW, pb, logitsb);
    softmax_kernel<<<BS, 256, 0, stream>>>(logitsb, wbuf);
    ctx_kernel<<<gb((size_t)BS * D), B256, 0, stream>>>(wbuf, enc_out, ctxbf);
    gru_step_dec<<<GRU_BLOCKS, B256, 0, stream>>>(
        hb_cur, ctxbf, decWhhB, decW1B, dec_Wih, dec_bih, dec_bhh,
        last, h_cur, h_nxt, hb_nxt);
    float* tf = h_cur;  h_cur = h_nxt;  h_nxt = tf;
    __bf16* tb16 = hb_cur; hb_cur = hb_nxt; hb_nxt = tb16;
    pred_kernel<<<BS, 256, 0, stream>>>(h_cur, tW, tb, out, last, t);
  }
  (void)in_sizes; (void)n_in; (void)out_size; (void)ws_size;
}
